// LRUMLP_layer_29566554865852
// MI455X (gfx1250) — matrix operand order repaired and checked by reading, not compiled, not run on hardware
//
#include <hip/hip_runtime.h>
#include <hip/hip_bf16.h>

typedef __attribute__((ext_vector_type(16))) __bf16 v16bf;
typedef __attribute__((ext_vector_type(8)))  float  v8f;

#define B_   16
#define T_   4096
#define TS_  64
#define DE_  128
#define H_   256
#define LD_  512
#define OD_  10
#define NBLK 6
#define MR   (B_ * T_)        // 65536 rows
#define NCH  32               // scan chunks
#define CLN  128              // chunk length (T_/NCH)

#define MTB  256              // rows per block (8 waves x 2 m-tiles x 16)
#define NT   64               // cols per block (4 n-tiles x 16)
#define KT   32

// epilogue flag bits
#define EF_BIAS  1
#define EF_ACC   2
#define EF_DTERM 4
#define EF_RES   8
#define EF_COPY2 16

// ---------------------------------------------------------------------------
// bf16-WMMA GEMM, no LDS: Out[M x N] (+=) A'[M x K] * BT^T
//   BT pre-transposed [N][K] bf16 (tiny, L2/WGP$-resident).
//   Each wave: 2 m-tiles x 4 n-tiles (B fragments reused across m-tiles).
//   All fragment addresses = hoisted base + compile-time immediate.
//   PRO=0: A' = A;  PRO=2: A' = GLU(A) = A[m,k]*sigmoid(A[m,k+256]), lda=512
// ---------------------------------------------------------------------------
template <int PRO, int EPI, int K>
__global__ __launch_bounds__(256) void gemm_wmma(
    const float* __restrict__ A, int lda,
    const __bf16* __restrict__ BT,           // [N][K] bf16
    float* __restrict__ Out, int N,
    const float* __restrict__ biasN,
    const float* __restrict__ resid,
    float* __restrict__ Out2,
    const float* __restrict__ hbuf,
    const float* __restrict__ dv1, const float* __restrict__ dv2)
{
    const int tid  = threadIdx.x;
    const int wv   = tid >> 5;
    const int lane = tid & 31;
    const int hx   = lane >> 4;
    const int l15  = lane & 15;
    const int m0   = blockIdx.x * MTB;
    const int n0   = blockIdx.y * NT;

    // hoisted bases: all inner-loop offsets are compile-time constants
    const float*  __restrict__ ar0 = A + (size_t)(m0 + wv * 32 + l15) * lda + hx * 16;
    const float*  __restrict__ ar1 = ar0 + (size_t)16 * lda;
    const __bf16* __restrict__ bb  = BT + (size_t)(n0 + l15) * K + hx * 16;

    v8f acc[2][4] = {};

#pragma unroll
    for (int k0 = 0; k0 < K; k0 += KT) {
        // B fragments for this K-step (shared by both m-tiles), immediate offsets
        v16bf bf[4];
#pragma unroll
        for (int nt = 0; nt < 4; ++nt)
            bf[nt] = *(const v16bf*)(bb + nt * 16 * K + k0);

        // A fragments: one contiguous 64B f32 segment per row
        v16bf af[2];
#pragma unroll
        for (int mt = 0; mt < 2; ++mt) {
            const float* ap = (mt == 0 ? ar0 : ar1) + k0;
            float4 a0 = *(const float4*)(ap);
            float4 a1 = *(const float4*)(ap + 4);
            float4 a2 = *(const float4*)(ap + 8);
            float4 a3 = *(const float4*)(ap + 12);
            float av[16] = { a0.x,a0.y,a0.z,a0.w, a1.x,a1.y,a1.z,a1.w,
                             a2.x,a2.y,a2.z,a2.w, a3.x,a3.y,a3.z,a3.w };
            if (PRO == 2) {
                const float* gp = ap + 256;
                float4 g0 = *(const float4*)(gp);
                float4 g1 = *(const float4*)(gp + 4);
                float4 g2 = *(const float4*)(gp + 8);
                float4 g3 = *(const float4*)(gp + 12);
                float gv[16] = { g0.x,g0.y,g0.z,g0.w, g1.x,g1.y,g1.z,g1.w,
                                 g2.x,g2.y,g2.z,g2.w, g3.x,g3.y,g3.z,g3.w };
#pragma unroll
                for (int e = 0; e < 16; ++e)
                    av[e] *= 1.0f / (1.0f + __expf(-gv[e]));
            }
            v16bf t;
#pragma unroll
            for (int e = 0; e < 16; ++e) t[e] = (__bf16)av[e];
            af[mt] = t;
        }

        // 8 independent WMMAs sharing operands
#pragma unroll
        for (int mt = 0; mt < 2; ++mt)
#pragma unroll
            for (int nt = 0; nt < 4; ++nt)
                acc[mt][nt] = __builtin_amdgcn_wmma_f32_16x16x32_bf16(
                    false, af[mt], false, bf[nt], (short)0, acc[mt][nt], false, false);
    }

    // epilogue: lane l15 = col, VGPR v -> row v + 8*hx (ISA C/D layout)
#pragma unroll
    for (int mt = 0; mt < 2; ++mt) {
#pragma unroll
        for (int nt = 0; nt < 4; ++nt) {
#pragma unroll
            for (int v = 0; v < 8; ++v) {
                int row = m0 + wv * 32 + mt * 16 + hx * 8 + v;
                int col = n0 + nt * 16 + l15;
                size_t off = (size_t)row * N + col;
                float d = acc[mt][nt][v];
                if (EPI & EF_BIAS)  d += biasN[col];
                if (EPI & EF_ACC)   d += Out[off];
                if (EPI & EF_DTERM) d += hbuf[(size_t)row * DE_ + col] * dv1[col] + dv2[col];
                if (EPI & EF_RES)   d += resid[off];
                Out[off] = d;
                if (EPI & EF_COPY2) Out2[off] = d;
            }
        }
    }
}

// ---------------------------------------------------------------------------
// BatchNorm statistics (two-stage deterministic reduction)
// ---------------------------------------------------------------------------
__global__ __launch_bounds__(256) void bn_partial(const float* __restrict__ h,
                                                  float* __restrict__ part)
{
    __shared__ float s1m[256], s2m[256];
    int t = threadIdx.x, b = blockIdx.x;
    int d = t & 127, seg = t >> 7;
    const float* base = h + (size_t)(b * 256 + seg * 128) * DE_ + d;
    float s = 0.f, q = 0.f;
    for (int r = 0; r < 128; ++r) { float v = base[(size_t)r * DE_]; s += v; q += v * v; }
    s1m[t] = s; s2m[t] = q;
    __syncthreads();
    if (t < 128) {
        s = s1m[t] + s1m[t + 128];
        q = s2m[t] + s2m[t + 128];
        part[(size_t)b * 256 + d * 2 + 0] = s;
        part[(size_t)b * 256 + d * 2 + 1] = q;
    }
}

__global__ void bn_final(const float* __restrict__ part,
                         float* __restrict__ mean, float* __restrict__ rs)
{
    int d = threadIdx.x;
    if (d >= 128) return;
    float s = 0.f, q = 0.f;
    for (int b = 0; b < 256; ++b) {
        s += part[(size_t)b * 256 + d * 2 + 0];
        q += part[(size_t)b * 256 + d * 2 + 1];
    }
    float m = s * (1.0f / (float)MR);
    float v = q * (1.0f / (float)MR) - m * m;
    mean[d] = m;
    rs[d]   = rsqrtf(v + 1e-5f);
}

// ---------------------------------------------------------------------------
// Static weight prep: bf16 transposes of W_enc, W1[i], W2[i]
// ---------------------------------------------------------------------------
__global__ void prep_weights(const float* __restrict__ W_enc,
                             const float* __restrict__ W1,
                             const float* __restrict__ W2,
                             __bf16* __restrict__ WencT,
                             __bf16* __restrict__ W1T,
                             __bf16* __restrict__ W2T)
{
    int gt = blockIdx.x * 256 + threadIdx.x;
    int tot = gridDim.x * 256;
    for (int idx = gt; idx < DE_ * TS_; idx += tot) {            // [128][64]
        int n = idx >> 6, k = idx & 63;
        WencT[idx] = (__bf16)W_enc[k * DE_ + n];
    }
    for (int idx = gt; idx < NBLK * LD_ * DE_; idx += tot) {     // 6 x [512][128]
        int b = idx >> 16, r = idx & 65535;
        int n = r >> 7, k = r & 127;
        W1T[idx] = (__bf16)W1[((size_t)b * DE_ + k) * LD_ + n];
    }
    for (int idx = gt; idx < NBLK * DE_ * (LD_/2); idx += tot) { // 6 x [128][256]
        int b = idx >> 15, r = idx & 32767;
        int n = r >> 8, k = r & 255;
        W2T[idx] = (__bf16)W2[((size_t)b * (LD_/2) + k) * DE_ + n];
    }
}

// ---------------------------------------------------------------------------
// Per-block small params: lambda, gamma, lambda^CLN, folded-BN vectors
// ---------------------------------------------------------------------------
__global__ void prep_small(const float* __restrict__ nu_log,
                           const float* __restrict__ theta_log,
                           const float* __restrict__ bn_scale,
                           const float* __restrict__ bn_bias,
                           const float* __restrict__ D_lru, int blk,
                           const float* __restrict__ bn_mean,
                           const float* __restrict__ bn_rs,
                           float* __restrict__ lam_re, float* __restrict__ lam_im,
                           float* __restrict__ lamP_re, float* __restrict__ lamP_im,
                           float* __restrict__ gam,
                           float* __restrict__ s1v, float* __restrict__ s2v,
                           float* __restrict__ dv1, float* __restrict__ dv2)
{
    int t = threadIdx.x;
    if (t < 128) {
        float a = bn_rs[t] * bn_scale[blk * DE_ + t];
        float b = bn_bias[blk * DE_ + t] - bn_mean[t] * a;
        s1v[t] = a; s2v[t] = b;
        float dvv = D_lru[blk * DE_ + t];
        dv1[t] = a * dvv; dv2[t] = b * dvv;
    }
    if (t < 256) {
        float nu = __expf(nu_log[blk * H_ + t]);
        float th = __expf(theta_log[blk * H_ + t]);
        float r  = __expf(-nu);
        float lr = r * __cosf(th), li = r * __sinf(th);
        lam_re[t] = lr; lam_im[t] = li;
        gam[t] = sqrtf(fmaxf(1.0f - r * r, 1e-8f));
        float pr = lr, pi = li;
        for (int s = 0; s < 7; ++s) { float nr = pr*pr - pi*pi, ni = 2.f*pr*pi; pr = nr; pi = ni; } // lam^128
        lamP_re[t] = pr; lamP_im[t] = pi;
    }
}

// ---------------------------------------------------------------------------
// Per-block matrix prep (grid-stride)
// ---------------------------------------------------------------------------
__global__ void prep_mats(const float* __restrict__ B_re, const float* __restrict__ B_im,
                          const float* __restrict__ C_re, const float* __restrict__ C_im,
                          int blk,
                          const float* __restrict__ gam,
                          const float* __restrict__ s1v, const float* __restrict__ s2v,
                          __bf16* __restrict__ BgRT, __bf16* __restrict__ BgIT,
                          float* __restrict__ cR, float* __restrict__ cI,
                          __bf16* __restrict__ CrTb, __bf16* __restrict__ CiNTb)
{
    int gt = blockIdx.x * 256 + threadIdx.x;
    int tot = gridDim.x * 256;
    for (int idx = gt; idx < H_ * DE_; idx += tot) {
        int hh = idx >> 7, d = idx & 127;
        float g = gam[hh] * s1v[d];
        size_t src = ((size_t)blk * H_ + hh) * DE_ + d;
        BgRT[idx] = (__bf16)(B_re[src] * g);
        BgIT[idx] = (__bf16)(B_im[src] * g);
    }
    for (int idx = gt; idx < DE_ * H_; idx += tot) {
        size_t src = (size_t)blk * DE_ * H_ + idx;
        CrTb[idx]  = (__bf16)(C_re[src]);
        CiNTb[idx] = (__bf16)(-C_im[src]);
    }
    for (int hh = gt; hh < H_; hh += tot) {
        float g = gam[hh], sr = 0.f, si = 0.f;
        size_t base = ((size_t)blk * H_ + hh) * DE_;
        for (int d = 0; d < DE_; ++d) {
            float s = s2v[d];
            sr += s * B_re[base + d];
            si += s * B_im[base + d];
        }
        cR[hh] = sr * g; cI[hh] = si * g;
    }
}

// ---------------------------------------------------------------------------
// Chunked complex scan: h[t] = lam*h[t-1] + Bu[t], in-place over Bu
// ---------------------------------------------------------------------------
__global__ __launch_bounds__(256) void scan1(
    const float* __restrict__ br, const float* __restrict__ bi,
    const float* __restrict__ lr_, const float* __restrict__ li_,
    float* __restrict__ carry)
{
    int gid = blockIdx.x * 256 + threadIdx.x;   // b*(NCH*H) + c*H + h
    int hh = gid & 255, c = (gid >> 8) & (NCH - 1), b = gid >> 13;
    float lr = lr_[hh], li = li_[hh];
    float sr = 0.f, si = 0.f;
    size_t base = ((size_t)b * T_ + c * CLN) * H_ + hh;
    for (int t = 0; t < CLN; ++t) {
        float xr = br[base + (size_t)t * H_];
        float xi = bi[base + (size_t)t * H_];
        float nr = lr * sr - li * si + xr;
        float ni = lr * si + li * sr + xi;
        sr = nr; si = ni;
    }
    carry[(size_t)gid * 2 + 0] = sr;
    carry[(size_t)gid * 2 + 1] = si;
}

__global__ void scan2(float* __restrict__ carry,
                      const float* __restrict__ lamP_re,
                      const float* __restrict__ lamP_im)
{
    int gid = blockIdx.x * 256 + threadIdx.x;   // b*H + h
    int hh = gid & 255, b = gid >> 8;
    float pr = lamP_re[hh], pi = lamP_im[hh];
    float fr = 0.f, fi = 0.f;
    for (int c = 0; c < NCH; ++c) {
        size_t idx = ((size_t)(b * NCH + c) * H_ + hh) * 2;
        float sr = carry[idx], si = carry[idx + 1];
        carry[idx] = fr; carry[idx + 1] = fi;   // carry-in for chunk c
        float nr = pr * fr - pi * fi + sr;
        float ni = pr * fi + pi * fr + si;
        fr = nr; fi = ni;
    }
}

__global__ __launch_bounds__(256) void scan3(
    float* __restrict__ br, float* __restrict__ bi,
    const float* __restrict__ lr_, const float* __restrict__ li_,
    const float* __restrict__ carry)
{
    int gid = blockIdx.x * 256 + threadIdx.x;
    int hh = gid & 255, c = (gid >> 8) & (NCH - 1), b = gid >> 13;
    float lr = lr_[hh], li = li_[hh];
    float sr = carry[(size_t)gid * 2 + 0];
    float si = carry[(size_t)gid * 2 + 1];
    size_t base = ((size_t)b * T_ + c * CLN) * H_ + hh;
    for (int t = 0; t < CLN; ++t) {
        size_t o = base + (size_t)t * H_;
        float xr = br[o], xi = bi[o];
        float nr = lr * sr - li * si + xr;
        float ni = lr * si + li * sr + xi;
        sr = nr; si = ni;
        br[o] = sr; bi[o] = si;
    }
}

// ---------------------------------------------------------------------------
// Mean-pool over tokens + output projection
// ---------------------------------------------------------------------------
__global__ void pool_kernel(const float* __restrict__ h, float* __restrict__ pooled)
{
    int gid = blockIdx.x * 256 + threadIdx.x;   // b*DE + d
    if (gid >= B_ * DE_) return;
    int d = gid & 127, b = gid >> 7;
    const float* base = h + (size_t)b * T_ * DE_ + d;
    float s = 0.f;
    for (int t = 0; t < T_; ++t) s += base[(size_t)t * DE_];
    pooled[gid] = s * (1.0f / (float)T_);
}

__global__ void out_kernel(const float* __restrict__ pooled,
                           const float* __restrict__ W_out,
                           const float* __restrict__ b_out,
                           float* __restrict__ out)
{
    int gid = threadIdx.x;
    if (gid >= B_ * OD_) return;
    int od = gid % OD_, b = gid / OD_;
    float s = b_out[od];
    for (int d = 0; d < DE_; ++d) s += pooled[b * DE_ + d] * W_out[d * OD_ + od];
    out[b * OD_ + od] = s;
}

// ---------------------------------------------------------------------------
extern "C" void kernel_launch(void* const* d_in, const int* in_sizes, int n_in,
                              void* d_out, int out_size, void* d_ws, size_t ws_size,
                              hipStream_t stream)
{
    (void)in_sizes; (void)n_in; (void)out_size; (void)ws_size;

    const float* x         = (const float*)d_in[0];
    const float* nu_log    = (const float*)d_in[1];
    const float* theta_log = (const float*)d_in[2];
    const float* B_re      = (const float*)d_in[3];
    const float* B_im      = (const float*)d_in[4];
    const float* C_re      = (const float*)d_in[5];
    const float* C_im      = (const float*)d_in[6];
    const float* D_lru     = (const float*)d_in[7];
    const float* W1        = (const float*)d_in[8];
    const float* b1        = (const float*)d_in[9];
    const float* W2        = (const float*)d_in[10];
    const float* b2        = (const float*)d_in[11];
    const float* bn_scale  = (const float*)d_in[12];
    const float* bn_bias   = (const float*)d_in[13];
    const float* W_enc     = (const float*)d_in[14];
    const float* b_enc     = (const float*)d_in[15];
    const float* W_out     = (const float*)d_in[16];
    const float* b_out     = (const float*)d_in[17];
    float* out             = (float*)d_out;

    char* ws = (char*)d_ws;
    size_t off = 0;
    auto carve = [&](size_t bytes) -> char* {
        char* p = ws + off;
        off = (off + bytes + 255) & ~(size_t)255;
        return p;
    };
    float*  y_enc  = (float*)carve((size_t)MR * DE_ * 4);
    float*  h      = (float*)carve((size_t)MR * DE_ * 4);
    float*  bu_re  = (float*)carve((size_t)MR * H_  * 4);
    float*  bu_im  = (float*)carve((size_t)MR * H_  * 4);
    float*  y_lru  = (float*)carve((size_t)MR * DE_ * 4);
    float*  t1     = (float*)carve((size_t)MR * LD_ * 4);
    __bf16* WencT  = (__bf16*)carve((size_t)DE_ * TS_ * 2);
    __bf16* W1T    = (__bf16*)carve((size_t)NBLK * LD_ * DE_ * 2);
    __bf16* W2T    = (__bf16*)carve((size_t)NBLK * DE_ * (LD_/2) * 2);
    __bf16* BgRT   = (__bf16*)carve((size_t)H_ * DE_ * 2);
    __bf16* BgIT   = (__bf16*)carve((size_t)H_ * DE_ * 2);
    __bf16* CrTb   = (__bf16*)carve((size_t)DE_ * H_ * 2);
    __bf16* CiNTb  = (__bf16*)carve((size_t)DE_ * H_ * 2);
    float*  cR     = (float*)carve(H_ * 4);
    float*  cI     = (float*)carve(H_ * 4);
    float*  lam_re = (float*)carve(H_ * 4);
    float*  lam_im = (float*)carve(H_ * 4);
    float*  lamP_re= (float*)carve(H_ * 4);
    float*  lamP_im= (float*)carve(H_ * 4);
    float*  gam    = (float*)carve(H_ * 4);
    float*  s1v    = (float*)carve(DE_ * 4);
    float*  s2v    = (float*)carve(DE_ * 4);
    float*  dv1    = (float*)carve(DE_ * 4);
    float*  dv2    = (float*)carve(DE_ * 4);
    float*  bn_mean= (float*)carve(DE_ * 4);
    float*  bn_rs  = (float*)carve(DE_ * 4);
    float*  part   = (float*)carve((size_t)256 * DE_ * 2 * 4);
    float*  carry  = (float*)carve((size_t)B_ * NCH * H_ * 2 * 4);
    float*  pooled = (float*)carve((size_t)B_ * DE_ * 4);

    const dim3 blk(256);

    prep_weights<<<64, blk, 0, stream>>>(W_enc, W1, W2, WencT, W1T, W2T);

    // encoder: h = y_enc = x @ W_enc + b_enc
    gemm_wmma<0, EF_BIAS | EF_COPY2, TS_><<<dim3(MR / MTB, DE_ / NT), blk, 0, stream>>>(
        x, TS_, WencT, h, DE_, b_enc, nullptr, y_enc, nullptr, nullptr, nullptr);

    for (int i = 0; i < NBLK; ++i) {
        bn_partial<<<256, blk, 0, stream>>>(h, part);
        bn_final<<<1, 128, 0, stream>>>(part, bn_mean, bn_rs);

        prep_small<<<1, blk, 0, stream>>>(nu_log, theta_log, bn_scale, bn_bias, D_lru, i,
                                          bn_mean, bn_rs, lam_re, lam_im, lamP_re, lamP_im,
                                          gam, s1v, s2v, dv1, dv2);
        prep_mats<<<64, blk, 0, stream>>>(B_re, B_im, C_re, C_im, i, gam, s1v, s2v,
                                          BgRT, BgIT, cR, cI, CrTb, CiNTb);

        // Bu = BN(h) @ (B*gamma)^T  with BN folded into BT + bias
        gemm_wmma<0, EF_BIAS, DE_><<<dim3(MR / MTB, H_ / NT), blk, 0, stream>>>(
            h, DE_, BgRT, bu_re, H_, cR, nullptr, nullptr, nullptr, nullptr, nullptr);
        gemm_wmma<0, EF_BIAS, DE_><<<dim3(MR / MTB, H_ / NT), blk, 0, stream>>>(
            h, DE_, BgIT, bu_im, H_, cI, nullptr, nullptr, nullptr, nullptr, nullptr);

        // complex scan (3-pass chunked), in-place
        scan1<<<(B_ * NCH * H_) / 256, blk, 0, stream>>>(bu_re, bu_im, lam_re, lam_im, carry);
        scan2<<<(B_ * H_) / 256, blk, 0, stream>>>(carry, lamP_re, lamP_im);
        scan3<<<(B_ * NCH * H_) / 256, blk, 0, stream>>>(bu_re, bu_im, lam_re, lam_im, carry);

        // y_lru = h_re@Cre^T - h_im@Cim^T + BN(h)*D (folded into dv1/dv2)
        gemm_wmma<0, 0, H_><<<dim3(MR / MTB, DE_ / NT), blk, 0, stream>>>(
            bu_re, H_, CrTb, y_lru, DE_, nullptr, nullptr, nullptr, nullptr, nullptr, nullptr);
        gemm_wmma<0, EF_ACC | EF_DTERM, H_><<<dim3(MR / MTB, DE_ / NT), blk, 0, stream>>>(
            bu_im, H_, CiNTb, y_lru, DE_, nullptr, nullptr, nullptr, h, dv1, dv2);

        // t1 = y_lru @ W1 + b1
        gemm_wmma<0, EF_BIAS, DE_><<<dim3(MR / MTB, LD_ / NT), blk, 0, stream>>>(
            y_lru, DE_, W1T + (size_t)i * LD_ * DE_, t1, LD_,
            b1 + i * LD_, nullptr, nullptr, nullptr, nullptr, nullptr);

        // h = GLU(t1) @ W2 + b2 + y_enc
        gemm_wmma<2, EF_BIAS | EF_RES, LD_/2><<<dim3(MR / MTB, DE_ / NT), blk, 0, stream>>>(
            t1, LD_, W2T + (size_t)i * DE_ * (LD_/2), h, DE_,
            b2 + i * DE_, y_enc, nullptr, nullptr, nullptr, nullptr);
    }

    pool_kernel<<<(B_ * DE_ + 255) / 256, blk, 0, stream>>>(h, pooled);
    out_kernel<<<1, blk, 0, stream>>>(pooled, W_out, b_out, out);
}
